// CausalSelfAttention_15796889714832
// MI455X (gfx1250) — matrix operand order repaired and checked by reading, not compiled, not run on hardware
//
#include <hip/hip_runtime.h>
#include <hip/hip_bf16.h>

// ---------------------------------------------------------------------------
// CDNA5 (gfx1250) fused causal self-attention, bf16 WMMA path.
// B=2, T=2048, C=1024, n_head=16, hd=64.
// ---------------------------------------------------------------------------

typedef __attribute__((ext_vector_type(16))) __bf16 v16bf;
typedef __attribute__((ext_vector_type(8)))  float  v8f;
typedef __attribute__((ext_vector_type(4)))  unsigned int u32x4;
typedef __attribute__((ext_vector_type(8)))  int          i32x8;
typedef __attribute__((ext_vector_type(4)))  int          i32x4;
typedef int v4i_vs __attribute__((vector_size(16)));   // matches async-LDS builtin param

#define SEQ_T   2048
#define DIM_C   1024
#define NHEAD   16
#define HD      64
#define ROWS_BT 4096   // B*T

// --- CDNA5 feature gates (fallbacks keep the kernel compiling everywhere) ---
#if defined(__gfx1250__) && __has_builtin(__builtin_amdgcn_global_load_async_to_lds_b128) && __has_builtin(__builtin_amdgcn_s_wait_asynccnt)
#define HAVE_ASYNC_LDS 1
#else
#define HAVE_ASYNC_LDS 0
#endif

#if defined(__gfx1250__) && __has_builtin(__builtin_amdgcn_tensor_load_to_lds) && __has_builtin(__builtin_amdgcn_s_wait_tensorcnt)
#define HAVE_TDM 1
#if __has_include(<hip/amd_detail/amd_gfx1250_TDM.h>)
#define TDM_6ARG 1
#else
#define TDM_6ARG 0
#endif
#else
#define HAVE_TDM 0
#endif

// Async global->LDS b128 operand casts: AS(1)/AS(3) pointers to 16B int vectors.
#define GASV4(p) ((__attribute__((address_space(1))) v4i_vs*)(void*)(p))
#define LASV4(p) ((__attribute__((address_space(3))) v4i_vs*)(void*)(p))

__device__ __forceinline__ unsigned short f32_to_bf16_rne(float f) {
    unsigned int u = __builtin_bit_cast(unsigned int, f);
    unsigned int r = u + 0x7FFFu + ((u >> 16) & 1u);
    return (unsigned short)(r >> 16);
}
__device__ __forceinline__ unsigned short f32_to_bf16_trunc(float f) {
    return (unsigned short)(__builtin_bit_cast(unsigned int, f) >> 16);
}
__device__ __forceinline__ __bf16 us_bf(unsigned short s) {
    return __builtin_bit_cast(__bf16, s);
}
__device__ __forceinline__ float bf_f32(unsigned short s) {
    unsigned int u = ((unsigned int)s) << 16;
    return __builtin_bit_cast(float, u);
}

// xor-butterfly reductions across the 16-lane half-wave groups using
// DS_SWIZZLE_B32 group-of-32 patterns (SWAPX1/2/4/8).
__device__ __forceinline__ float wred_max16(float v) {
    v = fmaxf(v, __builtin_bit_cast(float, __builtin_amdgcn_ds_swizzle(__builtin_bit_cast(int, v), 0x041F)));
    v = fmaxf(v, __builtin_bit_cast(float, __builtin_amdgcn_ds_swizzle(__builtin_bit_cast(int, v), 0x081F)));
    v = fmaxf(v, __builtin_bit_cast(float, __builtin_amdgcn_ds_swizzle(__builtin_bit_cast(int, v), 0x101F)));
    v = fmaxf(v, __builtin_bit_cast(float, __builtin_amdgcn_ds_swizzle(__builtin_bit_cast(int, v), 0x201F)));
    return v;
}
__device__ __forceinline__ float wred_sum16(float v) {
    v += __builtin_bit_cast(float, __builtin_amdgcn_ds_swizzle(__builtin_bit_cast(int, v), 0x041F));
    v += __builtin_bit_cast(float, __builtin_amdgcn_ds_swizzle(__builtin_bit_cast(int, v), 0x081F));
    v += __builtin_bit_cast(float, __builtin_amdgcn_ds_swizzle(__builtin_bit_cast(int, v), 0x101F));
    v += __builtin_bit_cast(float, __builtin_amdgcn_ds_swizzle(__builtin_bit_cast(int, v), 0x201F));
    return v;
}

// ---------------------------------------------------------------------------
// Kernel 1: fp32 -> bf16 conversion
// ---------------------------------------------------------------------------
__global__ __launch_bounds__(256) void cvt_f32_bf16(const float* __restrict__ in,
                                                    unsigned short* __restrict__ out,
                                                    int n) {
    int i = blockIdx.x * 256 + threadIdx.x;
    if (i < n) out[i] = f32_to_bf16_rne(in[i]);
}

// ---------------------------------------------------------------------------
// Kernel 2/5: bf16 WMMA GEMM. C[M,N] = A[M,K] @ B[K,N], fp32 accumulate.
// Block = 128 threads (4 waves). Block tile 64x128, K-step 32.
// A tile staged via async global->LDS b128 copies (ASYNCcnt) when available;
// B tile staged TRANSPOSED ([n][k]) so B-fragment reads are 2x ds_load_b128.
// ---------------------------------------------------------------------------
template <bool OUT_BF16>
__global__ __launch_bounds__(128) void gemm_bf16_wmma(
    const unsigned short* __restrict__ A,
    const unsigned short* __restrict__ Bm,
    void* __restrict__ Cout,
    int M, int N, int K)
{
    __shared__ unsigned int   lAu[64 * 16];     // A tile: 64 rows x 32 bf16
    __shared__ unsigned short lBt[128 * 32];    // B tile transposed: [n][k]

    const int tid  = threadIdx.x;
    const int lane = tid & 31;
    const int wave = tid >> 5;
    const int hf   = lane >> 4;
    const int l16  = lane & 15;
    const int m0   = blockIdx.y * 64;
    const int n0   = blockIdx.x * 128;

    v8f acc[8];
#pragma unroll
    for (int nc = 0; nc < 8; ++nc)
#pragma unroll
        for (int r = 0; r < 8; ++r) acc[nc][r] = 0.0f;

    const unsigned int* Agu = (const unsigned int*)A;   // row stride K/2 uints
    const unsigned int* Bgu = (const unsigned int*)Bm;  // row stride N/2 uints
    const int Ku = K >> 1, Nu = N >> 1;

    for (int k0 = 0; k0 < K; k0 += 32) {
        // Stage A tile (64x32 bf16): async global->LDS when available
#if HAVE_ASYNC_LDS
#pragma unroll
        for (int s = tid; s < 256; s += 128) {      // 256 x 16B segments
            int r = s >> 2, part = s & 3;
            __builtin_amdgcn_global_load_async_to_lds_b128(
                GASV4(Agu + (size_t)(m0 + r) * Ku + (k0 >> 1) + part * 4),
                LASV4(lAu + r * 16 + part * 4), 0, 0);
        }
#else
#pragma unroll
        for (int i = tid; i < 64 * 16; i += 128) {
            int r = i >> 4, c = i & 15;
            lAu[i] = Agu[(m0 + r) * Ku + (k0 >> 1) + c];
        }
#endif
        // Stage B tile (32x128 bf16 = 32x64 dwords), written transposed
#pragma unroll
        for (int i = tid; i < 32 * 64; i += 128) {
            int r  = i >> 6;          // k row within tile
            int cu = i & 63;          // dword column (2 n's)
            unsigned int w = Bgu[(k0 + r) * Nu + (n0 >> 1) + cu];
            lBt[(2 * cu + 0) * 32 + r] = (unsigned short)(w & 0xFFFFu);
            lBt[(2 * cu + 1) * 32 + r] = (unsigned short)(w >> 16);
        }
#if HAVE_ASYNC_LDS
        __builtin_amdgcn_s_wait_asynccnt(0);
#endif
        __syncthreads();

        const unsigned short* lA = (const unsigned short*)lAu;

        // A fragment: M = 16*wave + l16 ; K(e) = 16*hf + e
        v16bf af;
        {
            int Ml = wave * 16 + l16;
#pragma unroll
            for (int e = 0; e < 16; ++e) {
                int kk = hf * 16 + e;
                af[e] = us_bf(lA[Ml * 32 + kk]);
            }
        }
        // B fragments: N = 16*nc + l16 ; K(e) = 16*hf + e  (contiguous in lBt)
#pragma unroll
        for (int nc = 0; nc < 8; ++nc) {
            v16bf bfrag;
            int n = nc * 16 + l16;
#pragma unroll
            for (int e = 0; e < 16; ++e)
                bfrag[e] = us_bf(lBt[n * 32 + hf * 16 + e]);
            acc[nc] = __builtin_amdgcn_wmma_f32_16x16x32_bf16(
                false, af, false, bfrag, (short)0, acc[nc], false, false);
        }
        __syncthreads();
    }

    // Epilogue
#pragma unroll
    for (int nc = 0; nc < 8; ++nc)
#pragma unroll
        for (int r = 0; r < 8; ++r) {
            int row = m0 + wave * 16 + r + 8 * hf;
            int col = n0 + nc * 16 + l16;
            float v = acc[nc][r];
            if (OUT_BF16)
                ((unsigned short*)Cout)[(size_t)row * N + col] = f32_to_bf16_rne(v);
            else
                ((float*)Cout)[(size_t)row * N + col] = v;
        }
}

// ---------------------------------------------------------------------------
// Kernel 3: RoPE + pack qkv[4096,3072] (bf16) into q/k/v [B*nh, T, 64] (bf16)
// ---------------------------------------------------------------------------
__global__ __launch_bounds__(256) void rope_pack(
    const unsigned short* __restrict__ qkv,
    unsigned short* __restrict__ Qp,
    unsigned short* __restrict__ Kp,
    unsigned short* __restrict__ Vp)
{
    int idx = blockIdx.x * 256 + threadIdx.x;      // over 4096*1024
    if (idx >= ROWS_BT * DIM_C) return;
    int row = idx >> 10;          // b*T + t
    int c   = idx & 1023;         // channel
    int t   = row & (SEQ_T - 1);
    int b   = row >> 11;
    int h   = c >> 6;
    int d   = c & 63;

    const unsigned short* base = qkv + (size_t)row * (3 * DIM_C);
    float qv = bf_f32(base[c]);
    float kv = bf_f32(base[DIM_C + c]);
    float vv = bf_f32(base[2 * DIM_C + c]);

    int   dp = (d < 32) ? (c + 32) : (c - 32);
    float qpart = bf_f32(base[dp]);
    float kpart = bf_f32(base[DIM_C + dp]);
    float rq = (d < 32) ? -qpart : qpart;
    float rk = (d < 32) ? -kpart : kpart;

    int   i   = d & 31;
    float inv = __powf(10000.0f, -(float)(2 * i) / (float)HD);
    float th  = (float)t * inv;
    float sn, cs;
    __sincosf(th, &sn, &cs);

    float qo = qv * cs + rq * sn;
    float ko = kv * cs + rk * sn;

    size_t po = (((size_t)(b * NHEAD + h)) * SEQ_T + t) * HD + d;
    Qp[po] = f32_to_bf16_rne(qo);
    Kp[po] = f32_to_bf16_rne(ko);
    Vp[po] = f32_to_bf16_rne(vv);
}

// ---------------------------------------------------------------------------
// Kernel 4: flash attention (causal, online softmax), bf16 WMMA.
// grid = (B*nh, T/128), block = 256 threads (8 waves), 16KB dynamic LDS:
//   [0,     4096)  K tile  [key][dim] 32x64 bf16   (TDM / async target)
//   [4096,  8192)  V tile transposed [dim][key]
//   [8192, 16384)  per-wave P scratch (8 x 1KB)
// K tile is DMA'd by the Tensor Data Mover when available (TENSORcnt),
// else async global->LDS b128, else synchronous staging.
// ---------------------------------------------------------------------------
__global__ __launch_bounds__(256) void flash_attn_wmma(
    const unsigned short* __restrict__ Q,
    const unsigned short* __restrict__ Kp,
    const unsigned short* __restrict__ Vp,
    unsigned short* __restrict__ Y)
{
    extern __shared__ __align__(16) char smem[];
    unsigned int*   lK  = (unsigned int*)smem;                    // 4 KB
    unsigned short* lVt = (unsigned short*)(smem + 4096);         // 4 KB
    unsigned short* lP  = (unsigned short*)(smem + 8192);         // 8 KB

    const int tid  = threadIdx.x;
    const int lane = tid & 31;
    const int wave = tid >> 5;
    const int hf   = lane >> 4;
    const int l16  = lane & 15;

    const int bh   = blockIdx.x;     // b*16 + h
    const int tile = blockIdx.y;
    const int b    = bh >> 4;
    const int h    = bh & 15;

    const unsigned short* q  = Q  + (size_t)bh * SEQ_T * HD;
    const unsigned short* kg = Kp + (size_t)bh * SEQ_T * HD;
    const unsigned short* vg = Vp + (size_t)bh * SEQ_T * HD;

    const int qbase = tile * 128 + wave * 16;

#if HAVE_TDM
    const unsigned ldsK = (unsigned)(size_t)(__attribute__((address_space(3))) char*)smem;
#endif

    // Load Q fragments for dims [0,32) and [32,64): A layout.
    v16bf aq[2];
    {
        const unsigned short* qr = q + (size_t)(qbase + l16) * HD;
#pragma unroll
        for (int c = 0; c < 2; ++c)
#pragma unroll
            for (int e = 0; e < 16; ++e) {
                int d = c * 32 + hf * 16 + e;
                aq[c][e] = us_bf(qr[d]);
            }
    }

    float mrow[8], lrow[8];
    v8f   o[4];
#pragma unroll
    for (int r = 0; r < 8; ++r) { mrow[r] = -1e30f; lrow[r] = 0.0f; }
#pragma unroll
    for (int nc = 0; nc < 4; ++nc)
#pragma unroll
        for (int r = 0; r < 8; ++r) o[nc][r] = 0.0f;

    const float scale = 0.125f;            // 1/sqrt(64)
    const int   nkb   = (tile + 1) * 4;    // causal limit for the whole block

    for (int kb = 0; kb < nkb; ++kb) {
        const int kbase = kb * 32;

        // ---- Stage K tile (row-major 32x64 bf16) ----
#if HAVE_TDM
        if (tid < 32) {
            // 2D Tensor DMA descriptor (D#): data_size=2B, dim0=64(HD),
            // dim0_stride=64, tile = 64 x 32, dest = lK.
            size_t ga = (size_t)(const void*)(kg + (size_t)kbase * HD);
            u32x4 g0;
            g0[0] = 1u;                                         // count=1
            g0[1] = ldsK;                                       // lds_addr
            g0[2] = (unsigned)(ga & 0xFFFFFFFFu);               // global_addr lo
            g0[3] = (unsigned)((ga >> 32) & 0x1FFFFFFu) | (2u << 30); // hi | type=2
            i32x8 g1;
            g1[0] = (int)(1u << 16);          // data_size = 2 bytes
            g1[1] = (int)((unsigned)HD << 16);    // tensor_dim0 = 64
            g1[2] = (int)((unsigned)SEQ_T << 16); // tensor_dim1 = 2048
            g1[3] = (int)((unsigned)HD << 16);    // tile_dim0 = 64
            g1[4] = 32;                       // tile_dim1 = 32
            g1[5] = HD;                       // tensor_dim0_stride = 64
            g1[6] = 0;
            g1[7] = 0;
            i32x4 z4 = {0, 0, 0, 0};
#if TDM_6ARG
            i32x8 z8 = {0, 0, 0, 0, 0, 0, 0, 0};
            __builtin_amdgcn_tensor_load_to_lds(g0, g1, z4, z4, z8, 0);
#else
            __builtin_amdgcn_tensor_load_to_lds(g0, g1, z4, z4, 0);
#endif
            __builtin_amdgcn_s_wait_tensorcnt(0);
        }
#elif HAVE_ASYNC_LDS
        {   // 32 rows x 128B = 256 x 16B segments, one per thread
            int rk = tid >> 3, part = tid & 7;
            __builtin_amdgcn_global_load_async_to_lds_b128(
                GASV4(kg + (size_t)(kbase + rk) * HD + part * 8),
                LASV4((char*)lK + rk * 128 + part * 16), 0, 0);
        }
#else
#pragma unroll
        for (int i = tid; i < 32 * 32; i += 256) {
            int r = i >> 5, c = i & 31;
            lK[i] = ((const unsigned int*)kg)[(size_t)(kbase + r) * 32 + c];
        }
#endif

        // ---- Stage V tile transposed ([dim][key]) ----
#pragma unroll
        for (int i = tid; i < 32 * 32; i += 256) {
            int r = i >> 5, c = i & 31;
            unsigned int wv = ((const unsigned int*)vg)[(size_t)(kbase + r) * 32 + c];
            lVt[(2 * c + 0) * 32 + r] = (unsigned short)(wv & 0xFFFFu);
            lVt[(2 * c + 1) * 32 + r] = (unsigned short)(wv >> 16);
        }
        // Prefetch next K/V tile rows (global_prefetch_b8)
        if (wave == 0 && kb + 1 < nkb) {
            int nr = kbase + 32 + lane;
            __builtin_prefetch(kg + (size_t)nr * HD, 0, 0);
            __builtin_prefetch(vg + (size_t)nr * HD, 0, 0);
        }
#if HAVE_ASYNC_LDS && !HAVE_TDM
        __builtin_amdgcn_s_wait_asynccnt(0);
#endif
        __syncthreads();

        const unsigned short* lKs = (const unsigned short*)lK;

        // S = Q K^T for two 16-key sub-blocks (each: 2 WMMAs over hd=64)
        v8f s[2];
#pragma unroll
        for (int sub = 0; sub < 2; ++sub) {
#pragma unroll
            for (int r = 0; r < 8; ++r) s[sub][r] = 0.0f;
            int keyl = sub * 16 + l16;                 // local key = B-frag column
#pragma unroll
            for (int c = 0; c < 2; ++c) {
                v16bf bk;
#pragma unroll
                for (int e = 0; e < 16; ++e) {
                    int d = c * 32 + hf * 16 + e;      // B K-index -> head dim
                    bk[e] = us_bf(lKs[keyl * 64 + d]);
                }
                s[sub] = __builtin_amdgcn_wmma_f32_16x16x32_bf16(
                    false, aq[c], false, bk, (short)0, s[sub], false, false);
            }
        }

        // Causal mask + online softmax (row = qbase + r + 8*hf)
#pragma unroll
        for (int r = 0; r < 8; ++r) {
            int qrow = qbase + r + 8 * hf;
            float v0 = s[0][r] * scale;
            float v1 = s[1][r] * scale;
            if (kbase + l16 > qrow)       v0 = -1e30f;
            if (kbase + 16 + l16 > qrow)  v1 = -1e30f;

            float mx   = wred_max16(fmaxf(v0, v1));
            float mnew = fmaxf(mrow[r], mx);
            float corr = __expf(mrow[r] - mnew);
            mrow[r] = mnew;

            float p0 = __expf(v0 - mnew);
            float p1 = __expf(v1 - mnew);
            s[0][r] = p0;
            s[1][r] = p1;

            float sm = wred_sum16(p0 + p1);
            lrow[r] = lrow[r] * corr + sm;
#pragma unroll
            for (int nc = 0; nc < 4; ++nc) o[nc][r] *= corr;
        }

        // Relayout P (C-layout) -> A-layout through per-wave LDS scratch
        unsigned short* pw = lP + wave * 512;
#pragma unroll
        for (int sub = 0; sub < 2; ++sub)
#pragma unroll
            for (int r = 0; r < 8; ++r) {
                int Mr = r + 8 * hf;
                pw[Mr * 32 + sub * 16 + l16] = f32_to_bf16_trunc(s[sub][r]);
            }
        v16bf ap;
#pragma unroll
        for (int e = 0; e < 16; ++e) {
            int kk = hf * 16 + e;
            ap[e] = us_bf(pw[l16 * 32 + kk]);
        }

        // O += P @ V : four 16-dim chunks, K=32 keys per WMMA.
#pragma unroll
        for (int nc = 0; nc < 4; ++nc) {
            v16bf bv;
            int dim = nc * 16 + l16;
#pragma unroll
            for (int e = 0; e < 16; ++e)
                bv[e] = us_bf(lVt[dim * 32 + hf * 16 + e]);
            o[nc] = __builtin_amdgcn_wmma_f32_16x16x32_bf16(
                false, ap, false, bv, (short)0, o[nc], false, false);
        }
        __syncthreads();
    }

    // Epilogue: y[b, t, h*64 + d] bf16 (row-major [4096, 1024] for proj GEMM)
#pragma unroll
    for (int r = 0; r < 8; ++r) {
        float inv_l = __builtin_amdgcn_rcpf(lrow[r]);
        int trow = qbase + r + 8 * hf;
#pragma unroll
        for (int nc = 0; nc < 4; ++nc) {
            int d = nc * 16 + l16;
            float val = o[nc][r] * inv_l;
            Y[((size_t)(b * SEQ_T + trow)) * DIM_C + h * HD + d] = f32_to_bf16_rne(val);
        }
    }
}

// ---------------------------------------------------------------------------
// Host-side launch
// ---------------------------------------------------------------------------
extern "C" void kernel_launch(void* const* d_in, const int* in_sizes, int n_in,
                              void* d_out, int out_size, void* d_ws, size_t ws_size,
                              hipStream_t stream) {
    const float* x      = (const float*)d_in[0];   // [2,2048,1024]
    const float* w_attn = (const float*)d_in[1];   // [1024,3072]
    const float* w_proj = (const float*)d_in[2];   // [1024,1024]
    (void)in_sizes; (void)n_in; (void)out_size; (void)ws_size;

    char* ws = (char*)d_ws;
    size_t off = 0;
    unsigned short* x_bf    = (unsigned short*)(ws + off); off += (size_t)ROWS_BT * DIM_C * 2;       // 8 MB
    unsigned short* wa_bf   = (unsigned short*)(ws + off); off += (size_t)DIM_C * 3 * DIM_C * 2;     // 6 MB
    unsigned short* wp_bf   = (unsigned short*)(ws + off); off += (size_t)DIM_C * DIM_C * 2;         // 2 MB
    unsigned short* qkv_bf  = (unsigned short*)(ws + off); off += (size_t)ROWS_BT * 3 * DIM_C * 2;   // 25 MB
    unsigned short* q_pack  = (unsigned short*)(ws + off); off += (size_t)ROWS_BT * DIM_C * 2;       // 8 MB
    unsigned short* k_pack  = (unsigned short*)(ws + off); off += (size_t)ROWS_BT * DIM_C * 2;       // 8 MB
    unsigned short* v_pack  = (unsigned short*)(ws + off); off += (size_t)ROWS_BT * DIM_C * 2;       // 8 MB
    unsigned short* y_pack  = (unsigned short*)(ws + off); off += (size_t)ROWS_BT * DIM_C * 2;       // 8 MB

    // 1) fp32 -> bf16 conversions
    {
        int n = ROWS_BT * DIM_C;
        cvt_f32_bf16<<<(n + 255) / 256, 256, 0, stream>>>(x, x_bf, n);
        n = DIM_C * 3 * DIM_C;
        cvt_f32_bf16<<<(n + 255) / 256, 256, 0, stream>>>(w_attn, wa_bf, n);
        n = DIM_C * DIM_C;
        cvt_f32_bf16<<<(n + 255) / 256, 256, 0, stream>>>(w_proj, wp_bf, n);
    }

    // 2) QKV GEMM: [4096,1024] @ [1024,3072] -> bf16 [4096,3072]
    {
        dim3 grid(3 * DIM_C / 128, ROWS_BT / 64);
        gemm_bf16_wmma<true><<<grid, 128, 0, stream>>>(x_bf, wa_bf, qkv_bf,
                                                       ROWS_BT, 3 * DIM_C, DIM_C);
    }

    // 3) RoPE + head pack
    {
        int n = ROWS_BT * DIM_C;
        rope_pack<<<(n + 255) / 256, 256, 0, stream>>>(qkv_bf, q_pack, k_pack, v_pack);
    }

    // 4) Flash attention (16 KB dynamic LDS)
    {
        dim3 grid(2 * NHEAD, SEQ_T / 128);
        flash_attn_wmma<<<grid, 256, 16384, stream>>>(q_pack, k_pack, v_pack, y_pack);
    }

    // 5) Output projection: [4096,1024] @ [1024,1024] -> fp32 d_out
    {
        dim3 grid(DIM_C / 128, ROWS_BT / 64);
        gemm_bf16_wmma<false><<<grid, 128, 0, stream>>>(y_pack, wp_bf, d_out,
                                                        ROWS_BT, DIM_C, DIM_C);
    }
}
